// LinearSpatialGVA_5471788335639
// MI455X (gfx1250) — matrix operand order repaired and checked by reading, not compiled, not run on hardware
//
#include <hip/hip_runtime.h>

// ---------------------------------------------------------------------------
// Types
// ---------------------------------------------------------------------------
typedef __bf16 bf16;
typedef __attribute__((ext_vector_type(16))) __bf16 v16bf;
typedef __attribute__((ext_vector_type(8)))  __bf16 v8bf;
typedef __attribute__((ext_vector_type(4)))  __bf16 v4bf;
typedef __attribute__((ext_vector_type(8)))  float  v8f;

#define DEV static __device__ __forceinline__

// Problem constants (match reference)
static constexpr int Bb  = 8;
static constexpr int Cch = 512;
static constexpr int Nsp = 8192;
static constexpr int Hh  = 8;
static constexpr int Dd  = 64;
static constexpr int HID = 1024;
static constexpr int G   = Bb * Hh; // 64 batched heads

DEV float phi_f(float x) { return x > 0.f ? x + 1.f : __expf(x); } // elu(x)+1

// ---------------------------------------------------------------------------
// Async global->LDS staging (CDNA5 GLOBAL_LOAD_ASYNC_TO_LDS_B128, ASYNCcnt).
// Builtin exists on this toolchain; parameter type is a b128 (int4) pointer
// in AS1 (global) / AS3 (LDS). AS1 shares the bit pattern with generic
// global pointers; AS3 = low 32 bits of the flat LDS aperture address
// (ISA 10.2 per-aperture address calc), so integer casts are exact.
// ---------------------------------------------------------------------------
#if __has_builtin(__builtin_amdgcn_global_load_async_to_lds_b128) && \
    __has_builtin(__builtin_amdgcn_s_wait_asynccnt)
#define HAVE_ASYNC_LDS 1
#endif

typedef int v4i_gcc __attribute__((vector_size(16)));
typedef __attribute__((address_space(1))) v4i_gcc as1_b128;
typedef __attribute__((address_space(3))) v4i_gcc as3_b128;

DEV void copy16_to_lds(const bf16* g, bf16* l) {
#ifdef HAVE_ASYNC_LDS
  __builtin_amdgcn_global_load_async_to_lds_b128(
      (as1_b128*)(unsigned long long)g,
      (as3_b128*)(unsigned int)(unsigned long long)l, 0, 0);
#else
  *reinterpret_cast<v8bf*>(l) = *reinterpret_cast<const v8bf*>(g);
#endif
}

DEV void stage_fence() {
#ifdef HAVE_ASYNC_LDS
  __builtin_amdgcn_s_wait_asynccnt(0);
#endif
  __syncthreads();
}

// ---------------------------------------------------------------------------
// WMMA fragment helpers (CDNA5 16x16x32 bf16, wave32).
// A 16x32 (row-major LDS, K contiguous): lane L&15 = M row; two contiguous
//   16B chunks at K = 8*(L>>4) and 16 + 8*(L>>4)  -> 2 x ds_load_b128.
// B 32x16 from TRANSPOSED LDS tile Bt[n][k] (K contiguous): lane L&15 = N,
//   two contiguous 16B chunks at K = 8*(L>>4) and 16 + 8*(L>>4) -> 2 x ds_load_b128.
// ---------------------------------------------------------------------------
DEV v16bf load_afrag(const bf16* rowp, int lhalf) {
  union { v16bf v; v8bf b[2]; } f;
  f.b[0] = *reinterpret_cast<const v8bf*>(rowp + 8 * lhalf);
  f.b[1] = *reinterpret_cast<const v8bf*>(rowp + 16 + 8 * lhalf);
  return f.v;
}

DEV v16bf load_bfrag(const bf16* colp, int lhalf) { // colp = &Bt[n*ldp + k0]
  union { v16bf v; v8bf b[2]; } f;
  const bf16* p = colp + 8 * lhalf;
  f.b[0] = *reinterpret_cast<const v8bf*>(p);
  f.b[1] = *reinterpret_cast<const v8bf*>(p + 16);
  return f.v;
}

DEV v8f wmma_bf16(v16bf a, v16bf b, v8f c) {
  return __builtin_amdgcn_wmma_f32_16x16x32_bf16(
      /*neg_a=*/false, a, /*neg_b=*/false, b,
      /*c_mod=*/(short)0, c, /*reuse_a=*/false, /*reuse_b=*/false);
}

// ---------------------------------------------------------------------------
// Generic 128x128x32 bf16 WMMA GEMM, double-buffered async staging,
// fused epilogues:
//   mode 0: plain fp32 row-major store (FFN)
//   mode 1: qkv epilogue (+bias, phi on q/k, scatter to head layouts)
//   mode 2: proj epilogue (+bias, residual with x, dual fp32/bf16 store)
// ---------------------------------------------------------------------------
struct GemmArgs {
  const bf16* A; const bf16* Bm;
  long long sA, sB;        // batch strides (elements)
  int lda, ldb, K, mode;
  // mode 0
  float* outF; long long sOut; int ldc;
  // mode 1 / 2
  const float* bias;
  bf16* qphi; bf16* kphiT; bf16* vmat;         // mode 1
  const float* xin; float* xresF; bf16* xresB; // mode 2
};

static constexpr int LDA_P = 40;  // padded row (elements) for 32-wide K tiles

__global__ __launch_bounds__(256) void gemm128_kernel(GemmArgs p) {
  __shared__ bf16 As[2 * 128 * LDA_P];
  __shared__ bf16 Bt[2 * 128 * LDA_P]; // B^T: Bt[n][k]
  const int tid = threadIdx.x;
  const int lane = tid & 31, wave = tid >> 5;
  const int lhalf = lane >> 4, l16 = lane & 15;
  const int wm = wave >> 2;   // 0..1 -> 64-row slab
  const int wn = wave & 3;    // 0..3 -> 32-col slab
  const int z  = blockIdx.z;
  const int m0 = blockIdx.y * 128;
  const int n0 = blockIdx.x * 128;
  const bf16* Ag = p.A  + (long long)z * p.sA;
  const bf16* Bg = p.Bm + (long long)z * p.sB;

  v8f acc[4][2] = {};

  const int arow = tid >> 1, acol = (tid & 1) * 16;   // A: 128x32, 16 el/thread
  const int bk = (tid & 15) * 2, bn = (tid >> 4) * 8; // B: 2k x 8n per thread

  auto stage = [&](int buf, int kk) {
    bf16* Asb = &As[buf * 128 * LDA_P];
    bf16* Btb = &Bt[buf * 128 * LDA_P];
    // A tile: row-major copy (async path when available)
    const bf16* ag = Ag + (long long)(m0 + arow) * p.lda + kk + acol;
    copy16_to_lds(ag,     &Asb[arow * LDA_P + acol]);
    copy16_to_lds(ag + 8, &Asb[arow * LDA_P + acol + 8]);
    // B tile: register pair-interleave -> transposed store (K contiguous)
    union { v8bf v; unsigned short h[8]; } r0, r1;
    r0.v = *reinterpret_cast<const v8bf*>(Bg + (long long)(kk + bk)     * p.ldb + n0 + bn);
    r1.v = *reinterpret_cast<const v8bf*>(Bg + (long long)(kk + bk + 1) * p.ldb + n0 + bn);
#pragma unroll
    for (int i = 0; i < 8; ++i) {
      const unsigned int u = (unsigned int)r0.h[i] | ((unsigned int)r1.h[i] << 16);
      *reinterpret_cast<unsigned int*>(&Btb[(bn + i) * LDA_P + bk]) = u;
    }
    if (kk + 32 < p.K) { // pull tile t+2 toward L2 -> global_prefetch_b8
      __builtin_prefetch(ag + 32, 0, 1);
      __builtin_prefetch(Bg + (long long)(kk + 32 + bk) * p.ldb + n0 + bn, 0, 1);
    }
  };

  stage(0, 0);
  stage_fence();

  for (int k0 = 0, it = 0; k0 < p.K; k0 += 32, ++it) {
    const int buf = it & 1;
    if (k0 + 32 < p.K) stage(buf ^ 1, k0 + 32); // overlap next-tile staging
    const bf16* Asb = &As[buf * 128 * LDA_P];
    const bf16* Btb = &Bt[buf * 128 * LDA_P];

    v16bf af[4];
#pragma unroll
    for (int i = 0; i < 4; ++i)
      af[i] = load_afrag(&Asb[(wm * 64 + i * 16 + l16) * LDA_P], lhalf);
    v16bf bfr[2];
#pragma unroll
    for (int j = 0; j < 2; ++j)
      bfr[j] = load_bfrag(&Btb[(wn * 32 + j * 16 + l16) * LDA_P], lhalf);
#pragma unroll
    for (int i = 0; i < 4; ++i)
#pragma unroll
      for (int j = 0; j < 2; ++j)
        acc[i][j] = wmma_bf16(af[i], bfr[j], acc[i][j]);

    stage_fence();
  }

  // Epilogue. C layout: VGPR r -> M = r + 8*lhalf, lane&15 -> N.
#pragma unroll
  for (int i = 0; i < 4; ++i) {
#pragma unroll
    for (int j = 0; j < 2; ++j) {
#pragma unroll
      for (int r = 0; r < 8; ++r) {
        const int row = m0 + wm * 64 + i * 16 + r + 8 * lhalf;
        const int col = n0 + wn * 32 + j * 16 + l16;
        float val = acc[i][j][r];
        if (p.mode == 0) {
          p.outF[(long long)z * p.sOut + (long long)row * p.ldc + col] = val;
        } else if (p.mode == 1) {
          val += p.bias[col];
          const int region = col >> 9;    // 0:q 1:k 2:v
          const int c2 = col & 511;
          const int h = c2 >> 6, d = c2 & 63;
          const long long gg = (long long)(z * Hh + h);
          if (region == 0)
            p.qphi[(gg * Nsp + row) * 64 + d] = (bf16)phi_f(val);
          else if (region == 1)
            p.kphiT[(gg * 64 + d) * (long long)Nsp + row] = (bf16)phi_f(val);
          else
            p.vmat[(gg * Nsp + row) * 64 + d] = (bf16)val;
        } else { // proj + residual
          val += p.bias[col];
          const long long idx = ((long long)z * Cch + col) * Nsp + row;
          const float xr = p.xin[idx] + val;
          p.xresF[idx] = xr;
          p.xresB[idx] = (bf16)xr;
        }
      }
    }
  }
}

// ---------------------------------------------------------------------------
// S = k_phi^T @ v : per head g, [64 x 8192] @ [8192 x 64] -> fp32 [64 x 64]
// ---------------------------------------------------------------------------
__global__ __launch_bounds__(256) void attn_s_kernel(const bf16* kphiT, const bf16* vmat, float* S) {
  __shared__ bf16 As[2 * 64 * LDA_P];
  __shared__ bf16 Bt[2 * 64 * LDA_P]; // v^T tile: Bt[n][k]
  const int tid = threadIdx.x, lane = tid & 31, wave = tid >> 5;
  const int lhalf = lane >> 4, l16 = lane & 15;
  const int g = blockIdx.x;
  const int wm = wave & 3;    // m-tile 0..3
  const int wn2 = wave >> 2;  // 0..1 -> n-tiles wn2*2 + {0,1}
  const bf16* A  = kphiT + (long long)g * 64 * Nsp;
  const bf16* Bm = vmat  + (long long)g * Nsp * 64;
  v8f acc[2] = {};
  const int arow = tid >> 2, acol = (tid & 3) * 8;    // A: 64x32, 8 el/thread
  const int bk = (tid & 15) * 2, bn = (tid >> 4) * 4; // B: 2k x 4n per thread

  auto stage = [&](int buf, int kk) {
    bf16* Asb = &As[buf * 64 * LDA_P];
    bf16* Btb = &Bt[buf * 64 * LDA_P];
    copy16_to_lds(A + (long long)arow * Nsp + kk + acol, &Asb[arow * LDA_P + acol]);
    union { v4bf v; unsigned short h[4]; } r0, r1;
    r0.v = *reinterpret_cast<const v4bf*>(Bm + (long long)(kk + bk)     * 64 + bn);
    r1.v = *reinterpret_cast<const v4bf*>(Bm + (long long)(kk + bk + 1) * 64 + bn);
#pragma unroll
    for (int i = 0; i < 4; ++i) {
      const unsigned int u = (unsigned int)r0.h[i] | ((unsigned int)r1.h[i] << 16);
      *reinterpret_cast<unsigned int*>(&Btb[(bn + i) * LDA_P + bk]) = u;
    }
    if (kk + 32 < Nsp) __builtin_prefetch(A + (long long)arow * Nsp + kk + 32 + acol, 0, 1);
  };

  stage(0, 0);
  stage_fence();

  for (int k0 = 0, it = 0; k0 < Nsp; k0 += 32, ++it) {
    const int buf = it & 1;
    if (k0 + 32 < Nsp) stage(buf ^ 1, k0 + 32);
    const bf16* Asb = &As[buf * 64 * LDA_P];
    const bf16* Btb = &Bt[buf * 64 * LDA_P];
    const v16bf af = load_afrag(&Asb[(wm * 16 + l16) * LDA_P], lhalf);
#pragma unroll
    for (int j = 0; j < 2; ++j) {
      const v16bf bfr = load_bfrag(&Btb[((wn2 * 2 + j) * 16 + l16) * LDA_P], lhalf);
      acc[j] = wmma_bf16(af, bfr, acc[j]);
    }
    stage_fence();
  }
#pragma unroll
  for (int j = 0; j < 2; ++j)
#pragma unroll
    for (int r = 0; r < 8; ++r) {
      const int row = wm * 16 + r + 8 * lhalf;
      const int col = (wn2 * 2 + j) * 16 + l16;
      S[((long long)g * 64 + row) * 64 + col] = acc[j][r];
    }
}

// ---------------------------------------------------------------------------
// out = (q_phi @ S) / denom -> bf16 [B, N, C] (head-interleaved along C)
// ---------------------------------------------------------------------------
static constexpr int LDK64 = 72; // padded 64-wide K rows

__global__ __launch_bounds__(256) void attn_out_kernel(const bf16* qphi, const bf16* Sb,
                                                       const float* denom, bf16* outA) {
  __shared__ bf16 As[128 * LDK64];
  __shared__ bf16 Bt[64 * LDK64];  // S^T tile: Bt[n][k]
  const int tid = threadIdx.x, lane = tid & 31, wave = tid >> 5;
  const int lhalf = lane >> 4, l16 = lane & 15;
  const int g  = blockIdx.y;
  const int m0 = blockIdx.x * 128;
  const bf16* A  = qphi + ((long long)g * Nsp + m0) * 64;
  const bf16* Bm = Sb + (long long)g * 64 * 64;
  {
    const int row = tid >> 1, colb = (tid & 1) * 32; // A: 128x64, 32 el/thread
    const bf16* ag = A + (long long)row * 64 + colb;
    bf16* al = &As[row * LDK64 + colb];
    copy16_to_lds(ag,      al);
    copy16_to_lds(ag + 8,  al + 8);
    copy16_to_lds(ag + 16, al + 16);
    copy16_to_lds(ag + 24, al + 24);
    const int bk = (tid & 31) * 2, bn = (tid >> 5) * 8; // B: 2k x 8n per thread
    union { v8bf v; unsigned short h[8]; } r0, r1;
    r0.v = *reinterpret_cast<const v8bf*>(Bm + (long long)bk * 64 + bn);
    r1.v = *reinterpret_cast<const v8bf*>(Bm + (long long)(bk + 1) * 64 + bn);
#pragma unroll
    for (int i = 0; i < 8; ++i) {
      const unsigned int u = (unsigned int)r0.h[i] | ((unsigned int)r1.h[i] << 16);
      *reinterpret_cast<unsigned int*>(&Bt[(bn + i) * LDK64 + bk]) = u;
    }
  }
  stage_fence();

  v8f acc[4] = {};
#pragma unroll
  for (int ks = 0; ks < 2; ++ks) {
    const int k0 = ks * 32;
    const v16bf af = load_afrag(&As[(wave * 16 + l16) * LDK64 + k0], lhalf);
#pragma unroll
    for (int j = 0; j < 4; ++j) {
      const v16bf bfr = load_bfrag(&Bt[(j * 16 + l16) * LDK64 + k0], lhalf);
      acc[j] = wmma_bf16(af, bfr, acc[j]);
    }
  }
  const int b = g >> 3, h = g & 7;
#pragma unroll
  for (int j = 0; j < 4; ++j)
#pragma unroll
    for (int r = 0; r < 8; ++r) {
      const int row = m0 + wave * 16 + r + 8 * lhalf;  // spatial n
      const int col = j * 16 + l16;                    // d
      const float dn = denom[(long long)g * Nsp + row];
      outA[((long long)b * Nsp + row) * Cch + h * 64 + col] = (bf16)(acc[j][r] / dn);
    }
}

// ---------------------------------------------------------------------------
// Elementwise / reduction kernels
// ---------------------------------------------------------------------------
__global__ void cast_f32_bf16_kernel(const float* src, bf16* dst, long long n) {
  const long long i = (long long)blockIdx.x * 256 + threadIdx.x;
  if (i < n) dst[i] = (bf16)src[i];
}

__global__ __launch_bounds__(256) void transpose_cast_kernel(const float* x, bf16* xt) {
  __shared__ float tile[32][33];
  const int b = blockIdx.z;
  const int c0 = blockIdx.y * 32, n0 = blockIdx.x * 32;
  const int tx = threadIdx.x & 31, ty = threadIdx.x >> 5; // 32x8
#pragma unroll
  for (int i = 0; i < 32; i += 8)
    tile[ty + i][tx] = x[((long long)b * Cch + c0 + ty + i) * Nsp + n0 + tx];
  __syncthreads();
#pragma unroll
  for (int i = 0; i < 32; i += 8)
    xt[((long long)b * Nsp + n0 + ty + i) * Cch + c0 + tx] = (bf16)tile[tx][ty + i];
}

__global__ void ksum_kernel(const bf16* kphiT, float* ksum) {
  const int row = blockIdx.x * 8 + (threadIdx.x >> 5); // over G*64 rows
  const int lane = threadIdx.x & 31;
  const bf16* p = kphiT + (long long)row * Nsp;
  float s = 0.f;
  for (int i = lane; i < Nsp; i += 32) s += (float)p[i];
#pragma unroll
  for (int off = 16; off > 0; off >>= 1) s += __shfl_xor(s, off, 32);
  if (lane == 0) ksum[row] = s;
}

__global__ void denom_kernel(const bf16* qphi, const float* ksum, float* denom) {
  const long long idx = (long long)blockIdx.x * 256 + threadIdx.x; // over G*Nsp
  const int g = (int)(idx >> 13);  // /Nsp (8192 = 2^13)
  const bf16* qp = qphi + idx * 64;
  const float* ks = ksum + (long long)g * 64;
  float s = 1e-6f;
#pragma unroll
  for (int d = 0; d < 64; ++d) s += (float)qp[d] * ks[d];
  denom[idx] = s;
}

__global__ __launch_bounds__(256) void bn_stats_kernel(const float* t, const float* gamma,
                                                       const float* beta, float* scale,
                                                       float* shift, int CH) {
  const int ch = blockIdx.x;
  float s = 0.f, s2 = 0.f;
  for (int b = 0; b < Bb; ++b) {
    const float* p = t + ((long long)b * CH + ch) * Nsp;
    for (int i = threadIdx.x; i < Nsp; i += 256) { const float v = p[i]; s += v; s2 += v * v; }
  }
  __shared__ float rs[256], rq[256];
  rs[threadIdx.x] = s; rq[threadIdx.x] = s2;
  __syncthreads();
  for (int o = 128; o > 0; o >>= 1) {
    if (threadIdx.x < o) { rs[threadIdx.x] += rs[threadIdx.x + o]; rq[threadIdx.x] += rq[threadIdx.x + o]; }
    __syncthreads();
  }
  if (threadIdx.x == 0) {
    const float inv = 1.f / ((float)Bb * (float)Nsp);
    const float mean = rs[0] * inv;
    const float var = rq[0] * inv - mean * mean;
    const float sc = gamma[ch] * rsqrtf(var + 1e-5f);
    scale[ch] = sc; shift[ch] = beta[ch] - mean * sc;
  }
}

__global__ void bn_relu_cast_kernel(const float* t, const float* scale, const float* shift,
                                    bf16* o, int CHmask, long long total) {
  const long long i = (long long)blockIdx.x * 256 + threadIdx.x;
  if (i >= total) return;
  const int ch = (int)((i >> 13) & CHmask); // Nsp=2^13, CH power of two
  const float v = t[i] * scale[ch] + shift[ch];
  o[i] = (bf16)(v > 0.f ? v : 0.f);
}

__global__ void final_kernel(const float* xres, const float* h2, const float* scale,
                             const float* shift, float* out, long long total) {
  const long long i = (long long)blockIdx.x * 256 + threadIdx.x;
  if (i >= total) return;
  const int ch = (int)((i >> 13) & (Cch - 1));
  float v = h2[i] * scale[ch] + shift[ch];
  v = v > 0.f ? v : 0.f;
  out[i] = xres[i] + v;
}

// ---------------------------------------------------------------------------
// Launch
// ---------------------------------------------------------------------------
extern "C" void kernel_launch(void* const* d_in, const int* in_sizes, int n_in,
                              void* d_out, int out_size, void* d_ws, size_t ws_size,
                              hipStream_t stream) {
  (void)in_sizes; (void)n_in; (void)out_size; (void)ws_size;
  const float* x      = (const float*)d_in[0];
  const float* qkv_w  = (const float*)d_in[1];
  const float* qkv_b  = (const float*)d_in[2];
  const float* proj_w = (const float*)d_in[3];
  const float* proj_b = (const float*)d_in[4];
  const float* w1     = (const float*)d_in[5];
  const float* g1     = (const float*)d_in[6];
  const float* b1     = (const float*)d_in[7];
  const float* w2     = (const float*)d_in[8];
  const float* g2     = (const float*)d_in[9];
  const float* b2     = (const float*)d_in[10];
  float* out = (float*)d_out;

  char* ws = (char*)d_ws;
  size_t off = 0;
  auto alloc = [&](size_t bytes) -> void* {
    off = (off + 255) & ~(size_t)255;
    void* p = ws + off; off += bytes; return p;
  };

  const size_t EL_BNC = (size_t)Bb * Nsp * Cch;          // 33.5M
  bf16*  xt     = (bf16*)alloc(EL_BNC * 2);
  bf16*  wqkvB  = (bf16*)alloc((size_t)Cch * 3 * Cch * 2);
  bf16*  wprojB = (bf16*)alloc((size_t)Cch * Cch * 2);
  bf16*  w1B    = (bf16*)alloc((size_t)HID * Cch * 2);
  bf16*  w2B    = (bf16*)alloc((size_t)Cch * HID * 2);
  bf16*  qphi   = (bf16*)alloc(EL_BNC * 2);
  bf16*  kphiT  = (bf16*)alloc(EL_BNC * 2);
  bf16*  vmat   = (bf16*)alloc(EL_BNC * 2);
  float* S      = (float*)alloc((size_t)G * 64 * 64 * 4);
  bf16*  Sb     = (bf16*)alloc((size_t)G * 64 * 64 * 2);
  float* ksum   = (float*)alloc((size_t)G * 64 * 4);
  float* denomB = (float*)alloc((size_t)G * Nsp * 4);
  bf16*  outA   = (bf16*)alloc(EL_BNC * 2);
  float* xresF  = (float*)alloc(EL_BNC * 4);
  bf16*  xresB  = (bf16*)alloc(EL_BNC * 2);
  float* h1     = (float*)alloc((size_t)Bb * HID * Nsp * 4);
  bf16*  h1r    = (bf16*)alloc((size_t)Bb * HID * Nsp * 2);
  float* h2     = (float*)alloc(EL_BNC * 4);
  float* sc1    = (float*)alloc(HID * 4);
  float* sh1    = (float*)alloc(HID * 4);
  float* sc2    = (float*)alloc(Cch * 4);
  float* sh2    = (float*)alloc(Cch * 4);

  auto cast = [&](const float* s, bf16* d, long long n) {
    cast_f32_bf16_kernel<<<(unsigned)((n + 255) / 256), 256, 0, stream>>>(s, d, n);
  };

  // 0) weight casts + x transpose/cast
  cast(qkv_w,  wqkvB,  (long long)Cch * 3 * Cch);
  cast(proj_w, wprojB, (long long)Cch * Cch);
  cast(w1,     w1B,    (long long)HID * Cch);
  cast(w2,     w2B,    (long long)Cch * HID);
  transpose_cast_kernel<<<dim3(Nsp / 32, Cch / 32, Bb), 256, 0, stream>>>(x, xt);

  // 1) qkv GEMM: [N,C] @ [C,3C], fused bias + phi + head scatter
  {
    GemmArgs p = {};
    p.A = xt;     p.sA = (long long)Nsp * Cch; p.lda = Cch;
    p.Bm = wqkvB; p.sB = 0;                    p.ldb = 3 * Cch;
    p.K = Cch; p.mode = 1;
    p.bias = qkv_b; p.qphi = qphi; p.kphiT = kphiT; p.vmat = vmat;
    gemm128_kernel<<<dim3(3 * Cch / 128, Nsp / 128, Bb), 256, 0, stream>>>(p);
  }

  // 2) k_sum, S = k^T v, S->bf16, denom
  ksum_kernel<<<G * 64 / 8, 256, 0, stream>>>(kphiT, ksum);
  attn_s_kernel<<<G, 256, 0, stream>>>(kphiT, vmat, S);
  cast(S, Sb, (long long)G * 64 * 64);
  denom_kernel<<<(unsigned)((long long)G * Nsp / 256), 256, 0, stream>>>(qphi, ksum, denomB);

  // 3) out_attn = (q_phi @ S) / denom
  attn_out_kernel<<<dim3(Nsp / 128, G), 256, 0, stream>>>(qphi, Sb, denomB, outA);

  // 4) proj GEMM + residual (dual fp32/bf16 stores of x + y)
  {
    GemmArgs p = {};
    p.A = outA;    p.sA = (long long)Nsp * Cch; p.lda = Cch;
    p.Bm = wprojB; p.sB = 0;                    p.ldb = Cch;
    p.K = Cch; p.mode = 2;
    p.bias = proj_b; p.xin = x; p.xresF = xresF; p.xresB = xresB;
    gemm128_kernel<<<dim3(Cch / 128, Nsp / 128, Bb), 256, 0, stream>>>(p);
  }

  // 5) FFN1: h1 = w1 @ xres  ([HID,C] @ [C,N] per batch)
  {
    GemmArgs p = {};
    p.A = w1B;    p.sA = 0;                     p.lda = Cch;
    p.Bm = xresB; p.sB = (long long)Cch * Nsp;  p.ldb = Nsp;
    p.K = Cch; p.mode = 0;
    p.outF = h1; p.sOut = (long long)HID * Nsp; p.ldc = Nsp;
    gemm128_kernel<<<dim3(Nsp / 128, HID / 128, Bb), 256, 0, stream>>>(p);
  }
  bn_stats_kernel<<<HID, 256, 0, stream>>>(h1, g1, b1, sc1, sh1, HID);
  bn_relu_cast_kernel<<<(unsigned)((long long)Bb * HID * Nsp / 256), 256, 0, stream>>>(
      h1, sc1, sh1, h1r, HID - 1, (long long)Bb * HID * Nsp);

  // 6) FFN2: h2 = w2 @ relu(bn(h1))  ([C,HID] @ [HID,N] per batch)
  {
    GemmArgs p = {};
    p.A = w2B;  p.sA = 0;                       p.lda = HID;
    p.Bm = h1r; p.sB = (long long)HID * Nsp;    p.ldb = Nsp;
    p.K = HID; p.mode = 0;
    p.outF = h2; p.sOut = (long long)Cch * Nsp; p.ldc = Nsp;
    gemm128_kernel<<<dim3(Nsp / 128, Cch / 128, Bb), 256, 0, stream>>>(p);
  }
  bn_stats_kernel<<<Cch, 256, 0, stream>>>(h2, g2, b2, sc2, sh2, Cch);

  // 7) out = xres + relu(bn(h2))
  final_kernel<<<(unsigned)((long long)Bb * Cch * Nsp / 256), 256, 0, stream>>>(
      xresF, h2, sc2, sh2, out, (long long)Bb * Cch * Nsp);
}
